// GNNTrafficPredictor_83124797046831
// MI455X (gfx1250) — matrix operand order repaired and checked by reading, not compiled, not run on hardware
//
#include <hip/hip_runtime.h>
#include <hip/hip_bf16.h>

#define NN 100000
#define EE 1600000
#define FN 32
#define FE 8
#define HD 64
#define KP1 160   // 2H+Fe=136 padded to 5 chunks of 32

typedef __attribute__((ext_vector_type(16))) _Float16 v16h;
typedef __attribute__((ext_vector_type(8)))  _Float16 v8h;
typedef __attribute__((ext_vector_type(8)))  float    v8f;
typedef __attribute__((ext_vector_type(4)))  float    v4f;

__device__ __forceinline__ void set8(v16h& a, int o, const _Float16* p) {
  v8h v = *(const v8h*)p;
#pragma unroll
  for (int i = 0; i < 8; ++i) a[o + i] = v[i];
}

__device__ __forceinline__ v8f wmma16(v16h a, v16h b, v8f c) {
  return __builtin_amdgcn_wmma_f32_16x16x32_f16(false, a, false, b, (short)0, c, false, false);
}

__device__ __forceinline__ void fatomic_add(float* p, float v) {
  __hip_atomic_fetch_add(p, v, __ATOMIC_RELAXED, __HIP_MEMORY_SCOPE_AGENT);
}

// ---------- weight prep: transpose to [N][K] f16 (B-fragment friendly) ----------
__global__ void prep_weights(const float* We, const float* Wc1, const float* Wc2,
                             const float* Wp1,
                             _Float16* WtE, _Float16* Wt1, _Float16* Wt2, _Float16* WtP1) {
  int idx = blockIdx.x * blockDim.x + threadIdx.x;
  if (idx < HD * FN) {                       // 2048: WtE[n][k] = We[k][n]
    int n = idx / FN, k = idx % FN;
    WtE[idx] = (_Float16)We[k * HD + n];
  }
  if (idx < HD * HD) {                       // 4096
    int n = idx / HD, k = idx % HD;
    Wt1[idx] = (_Float16)Wc1[k * HD + n];
    Wt2[idx] = (_Float16)Wc2[k * HD + n];
  }
  if (idx < HD * KP1) {                      // 10240, zero-pad K>=136
    int n = idx / KP1, k = idx % KP1;
    WtP1[idx] = (k < 2 * HD + FE) ? (_Float16)Wp1[k * HD + n] : (_Float16)0.0f;
  }
}

__global__ void convert_x(const float* x, _Float16* xf, int n) {
  int i = blockIdx.x * blockDim.x + threadIdx.x;
  if (i < n) xf[i] = (_Float16)x[i];
}

__global__ void deg_init(float* deg, int n) {
  int i = blockIdx.x * blockDim.x + threadIdx.x;
  if (i < n) deg[i] = 1.0f;                  // self-loop
}

__global__ void deg_accum(const long long* ei, float* deg, int e) {
  int i = blockIdx.x * blockDim.x + threadIdx.x;
  if (i < e) fatomic_add(&deg[ei[(long long)EE + i]], 1.0f);
}

__global__ void dinv_kernel(const float* deg, float* dinv, int n) {
  int i = blockIdx.x * blockDim.x + threadIdx.x;
  if (i < n) dinv[i] = rsqrtf(deg[i]);
}

// ---------- embed: h = relu(x @ We + bE), f16 out. wave per 16-node tile ----------
__global__ void embed_wmma(const _Float16* xf, const _Float16* WtE, const float* bE,
                           _Float16* hout, int ntiles) {
  int wid = threadIdx.x >> 5, lane = threadIdx.x & 31;
  int tile = blockIdx.x * (blockDim.x >> 5) + wid;
  if (tile >= ntiles) return;                // wave-uniform: EXEC stays all-ones
  int n = lane & 15, half = lane >> 4;
  const _Float16* row = xf + (size_t)(tile * 16 + n) * FN;
  v16h a = {};
  set8(a, 0, row + half * 16);
  set8(a, 8, row + half * 16 + 8);
#pragma unroll
  for (int t = 0; t < 4; ++t) {
    v16h b = *(const v16h*)(WtE + (size_t)(t * 16 + n) * FN + 16 * half);
    v8f acc = {};
    acc = wmma16(a, b, acc);
    int col = t * 16 + n;
    float bias = bE[col];
#pragma unroll
    for (int r = 0; r < 8; ++r) {
      int m = r + 8 * half;
      float v = acc[r] + bias;
      v = v > 0.0f ? v : 0.0f;
      hout[(size_t)(tile * 16 + m) * HD + col] = (_Float16)v;
    }
  }
}

// ---------- dense GCN matmul: hw = h @ Wc (no bias/relu), f32 out ----------
__global__ void node_matmul_wmma(const _Float16* hin, const _Float16* Wt,
                                 float* hw, int ntiles) {
  int wid = threadIdx.x >> 5, lane = threadIdx.x & 31;
  int tile = blockIdx.x * (blockDim.x >> 5) + wid;
  if (tile >= ntiles) return;
  int n = lane & 15, half = lane >> 4;
  const _Float16* row = hin + (size_t)(tile * 16 + n) * HD;
  v16h a[2];
#pragma unroll
  for (int c = 0; c < 2; ++c) {
    v16h af = {};
    set8(af, 0, row + 32 * c + half * 16);
    set8(af, 8, row + 32 * c + half * 16 + 8);
    a[c] = af;
  }
#pragma unroll
  for (int t = 0; t < 4; ++t) {
    v8f acc = {};
#pragma unroll
    for (int c = 0; c < 2; ++c) {
      v16h b = *(const v16h*)(Wt + (size_t)(t * 16 + n) * HD + 32 * c + 16 * half);
      acc = wmma16(a[c], b, acc);
    }
    int col = t * 16 + n;
#pragma unroll
    for (int r = 0; r < 8; ++r) {
      int m = r + 8 * half;
      hw[(size_t)(tile * 16 + m) * HD + col] = acc[r];
    }
  }
}

// ---------- aggregation: acc = b + hw*dinv^2 (self loop) ----------
__global__ void agg_init(const float* hw, const float* dinv, const float* bias,
                         float* acc, int total) {
  int i = blockIdx.x * blockDim.x + threadIdx.x;
  if (i >= total) return;
  int node = i >> 6, j = i & 63;
  float di = dinv[node];
  acc[i] = bias[j] + hw[i] * di * di;
}

// ---------- aggregation: acc[dst] += hw[src] * dinv[s]*dinv[d], per (edge,feat) ----------
__global__ void agg_edges(const float* hw, const float* dinv, const long long* ei,
                          float* acc) {
  int i = blockIdx.x * blockDim.x + threadIdx.x;   // < EE*64
  int e = i >> 6, j = i & 63;
  long long s = ei[e];
  long long d = ei[(long long)EE + e];
  float nm = dinv[s] * dinv[d];
  fatomic_add(&acc[(size_t)d * HD + j], hw[(size_t)s * HD + j] * nm);
}

__global__ void relu_conv(const float* acc, _Float16* hout, int total) {
  int i = blockIdx.x * blockDim.x + threadIdx.x;
  if (i >= total) return;
  float v = acc[i];
  hout[i] = (_Float16)(v > 0.0f ? v : 0.0f);
}

// ---------- edge MLP: out[e] = relu([h[s],h[d],ea] @ Wp1 + b1) . w2 + b2 ----------
// wave per 16-edge tile; A-fragments gathered directly, z stays in registers,
// final matvec via shfl_xor butterfly over the 16 N-lanes.
__global__ void edge_mlp_wmma(const _Float16* h2, const long long* ei, const float* ea,
                              const _Float16* WtP1, const float* b1,
                              const float* w2, const float* b2,
                              float* out, int etiles) {
  int wid = threadIdx.x >> 5, lane = threadIdx.x & 31;
  int tile = blockIdx.x * (blockDim.x >> 5) + wid;
  if (tile >= etiles) return;
  int n = lane & 15, half = lane >> 4;
  long long e = (long long)tile * 16 + n;          // A-row for this lane
  long long s = ei[e];
  long long d = ei[(long long)EE + e];
  const _Float16* hs = h2 + (size_t)s * HD;
  const _Float16* hd = h2 + (size_t)d * HD;

  v16h a[5];
#pragma unroll
  for (int c = 0; c < 5; ++c) {
    v16h af = {};
#pragma unroll
    for (int run = 0; run < 2; ++run) {
      int kb = 32 * c + 16 * half + 8 * run;       // 8-aligned; never straddles 64/128
      if (kb < 64) {
        set8(af, run * 8, hs + kb);
      } else if (kb < 128) {
        set8(af, run * 8, hd + (kb - 64));
      } else if (kb < 136) {                       // kb==128: the 8 edge attrs (f32->f16)
        const float* p = ea + (size_t)e * FE;
        v4f lo = *(const v4f*)p;
        v4f hi = *(const v4f*)(p + 4);
#pragma unroll
        for (int i = 0; i < 4; ++i) {
          af[run * 8 + i]     = (_Float16)lo[i];
          af[run * 8 + 4 + i] = (_Float16)hi[i];
        }
      } // else: zero pad (K in [136,160))
    }
    a[c] = af;
  }

  float p[8] = {0.f, 0.f, 0.f, 0.f, 0.f, 0.f, 0.f, 0.f};
#pragma unroll
  for (int t = 0; t < 4; ++t) {
    v8f acc = {};
#pragma unroll
    for (int c = 0; c < 5; ++c) {
      v16h b = *(const v16h*)(WtP1 + (size_t)(t * 16 + n) * KP1 + 32 * c + 16 * half);
      acc = wmma16(a[c], b, acc);
    }
    int col = t * 16 + n;
    float bias = b1[col], wv = w2[col];
#pragma unroll
    for (int r = 0; r < 8; ++r) {
      float z = acc[r] + bias;                     // z[M=r+8*half][col]
      z = z > 0.0f ? z : 0.0f;
      p[r] += z * wv;                              // partial dot with w2
    }
  }
  // reduce over the 16 N-lanes (bit4 = half stays fixed)
#pragma unroll
  for (int m = 1; m < 16; m <<= 1) {
#pragma unroll
    for (int r = 0; r < 8; ++r) p[r] += __shfl_xor(p[r], m, 32);
  }
  if (n == 0) {
    float bb = b2[0];
#pragma unroll
    for (int r = 0; r < 8; ++r)
      out[(size_t)tile * 16 + 8 * half + r] = p[r] + bb;
  }
}

extern "C" void kernel_launch(void* const* d_in, const int* in_sizes, int n_in,
                              void* d_out, int out_size, void* d_ws, size_t ws_size,
                              hipStream_t stream) {
  (void)in_sizes; (void)n_in; (void)out_size; (void)ws_size;
  const float*     x   = (const float*)d_in[0];
  const float*     ea  = (const float*)d_in[1];
  const float*     We  = (const float*)d_in[2];
  const float*     bE  = (const float*)d_in[3];
  const float*     Wc1 = (const float*)d_in[4];
  const float*     bc1 = (const float*)d_in[5];
  const float*     Wc2 = (const float*)d_in[6];
  const float*     bc2 = (const float*)d_in[7];
  const float*     Wp1 = (const float*)d_in[8];
  const float*     bp1 = (const float*)d_in[9];
  const float*     Wp2 = (const float*)d_in[10];
  const float*     bp2 = (const float*)d_in[11];
  const long long* ei  = (const long long*)d_in[12];
  float* out = (float*)d_out;

  char* ws = (char*)d_ws;
  size_t off = 0;
  auto walloc = [&](size_t bytes) -> void* {
    void* p = ws + off;
    off += (bytes + 255) & ~(size_t)255;
    return p;
  };
  _Float16* xf16 = (_Float16*)walloc((size_t)NN * FN * 2);   //  6.4 MB
  _Float16* hf16 = (_Float16*)walloc((size_t)NN * HD * 2);   // 12.8 MB
  float*    hw   = (float*)   walloc((size_t)NN * HD * 4);   // 25.6 MB
  float*    acc  = (float*)   walloc((size_t)NN * HD * 4);   // 25.6 MB
  float*    deg  = (float*)   walloc((size_t)NN * 4);
  float*    dinv = (float*)   walloc((size_t)NN * 4);
  _Float16* WtE  = (_Float16*)walloc((size_t)HD * FN * 2);
  _Float16* Wt1  = (_Float16*)walloc((size_t)HD * HD * 2);
  _Float16* Wt2  = (_Float16*)walloc((size_t)HD * HD * 2);
  _Float16* WtP1 = (_Float16*)walloc((size_t)HD * KP1 * 2);

  const int ntiles = NN / 16;   // 6250
  const int etiles = EE / 16;   // 100000

  prep_weights<<<40, 256, 0, stream>>>(We, Wc1, Wc2, Wp1, WtE, Wt1, Wt2, WtP1);
  convert_x<<<(NN * FN + 255) / 256, 256, 0, stream>>>(x, xf16, NN * FN);
  deg_init<<<(NN + 255) / 256, 256, 0, stream>>>(deg, NN);
  deg_accum<<<(EE + 255) / 256, 256, 0, stream>>>(ei, deg, EE);
  dinv_kernel<<<(NN + 255) / 256, 256, 0, stream>>>(deg, dinv, NN);

  embed_wmma<<<(ntiles + 3) / 4, 128, 0, stream>>>(xf16, WtE, bE, hf16, ntiles);

  // GCN layer 1
  node_matmul_wmma<<<(ntiles + 3) / 4, 128, 0, stream>>>(hf16, Wt1, hw, ntiles);
  agg_init<<<(NN * HD + 255) / 256, 256, 0, stream>>>(hw, dinv, bc1, acc, NN * HD);
  agg_edges<<<(EE * HD) / 256, 256, 0, stream>>>(hw, dinv, ei, acc);
  relu_conv<<<(NN * HD + 255) / 256, 256, 0, stream>>>(acc, hf16, NN * HD);

  // GCN layer 2
  node_matmul_wmma<<<(ntiles + 3) / 4, 128, 0, stream>>>(hf16, Wt2, hw, ntiles);
  agg_init<<<(NN * HD + 255) / 256, 256, 0, stream>>>(hw, dinv, bc2, acc, NN * HD);
  agg_edges<<<(EE * HD) / 256, 256, 0, stream>>>(hw, dinv, ei, acc);
  relu_conv<<<(NN * HD + 255) / 256, 256, 0, stream>>>(acc, hf16, NN * HD);

  // edge MLP + final matvec, fused
  edge_mlp_wmma<<<etiles / 4, 128, 0, stream>>>(hf16, ei, ea, WtP1, bp1, Wp2, bp2,
                                                out, etiles);
}
